// PCNDetector_76519137345579
// MI455X (gfx1250) — compile-verified, matrix operand order checked by reading, not run on hardware
//
#include <hip/hip_runtime.h>
#include <hip/hip_bf16.h>
#include <stdint.h>

typedef __attribute__((ext_vector_type(16))) _Float16 v16h;
typedef __attribute__((ext_vector_type(8)))  float    v8f;

// ---------------------------------------------------------------------------
// Problem constants
// ---------------------------------------------------------------------------
#define BB   8
#define NN   4096
#define MM1  2048
#define MM2  512
#define KK   32
#define DD1  64
#define DD2  128
#define DD3  256
#define NROWS 30
#define NCOLS 40
#define NCELLS (NROWS*NCOLS)

// ---------------------------------------------------------------------------
// Weight packing into WMMA B-fragment layout (f16 pairs per lane)
// layout: pw[((ct*kChunks + ch)*32 + lane)*8 + slot]  (u32 = 2 packed f16)
// lane: n = ct*16 + (lane&15); hi = lane>>4
// slot i<4 : kl = 2*i      + 8*hi ;  slot i>=4 : kl = 16 + 2*(i-4) + 8*hi
// ---------------------------------------------------------------------------
__global__ __launch_bounds__(256) void pack_w_kernel(const float* __restrict__ W,
                                                     int fin, int fout,
                                                     uint32_t* __restrict__ out,
                                                     int kChunks) {
  int t = blockIdx.x * 256 + threadIdx.x;
  int colTiles = fout >> 4;
  int total = colTiles * kChunks * 32;
  if (t >= total) return;
  int lane  = t & 31;
  int chunk = (t >> 5) % kChunks;
  int ct    = (t >> 5) / kChunks;
  int n  = ct * 16 + (lane & 15);
  int hi = lane >> 4;
#pragma unroll
  for (int i = 0; i < 8; ++i) {
    int kl = (i < 4) ? (2 * i + 8 * hi) : (16 + 2 * (i - 4) + 8 * hi);
    int k = chunk * 32 + kl;
    float a = (k     < fin) ? W[(size_t)k * fout + n]       : 0.0f;
    float b = (k + 1 < fin) ? W[(size_t)(k + 1) * fout + n] : 0.0f;
    union { _Float16 h[2]; uint32_t u; } cv;
    cv.h[0] = (_Float16)a; cv.h[1] = (_Float16)b;
    out[(size_t)t * 8 + i] = cv.u;
  }
}

// ---------------------------------------------------------------------------
// One 16x16 output tile: A from LDS (row-major f16, strideH halves),
// B from pre-packed global fragments.  Accumulate over kChunks in f32.
// ---------------------------------------------------------------------------
__device__ inline v8f tile_acc(const uint32_t* A32, int strideH, int rowBase,
                               const uint32_t* __restrict__ pw, int kChunks, int ct) {
  int lane = threadIdx.x & 31;
  int mr = lane & 15, hi = lane >> 4;
  v8f acc = {};
  int rowOff = (rowBase + mr) * strideH;
  for (int ch = 0; ch < kChunks; ++ch) {
    union { v16h v; uint32_t u[8]; } af;
    int kb = ch * 32 + 8 * hi;
#pragma unroll
    for (int i = 0; i < 8; ++i) {
      int kl = (i < 4) ? (2 * i) : (16 + 2 * (i - 4));
      af.u[i] = A32[(rowOff + kb + kl) >> 1];
    }
    union { v16h v; uint4 q[2]; } bf;
    const uint4* bp = (const uint4*)(pw + (((size_t)ct * kChunks + ch) * 32 + lane) * 8);
    bf.q[0] = bp[0]; bf.q[1] = bp[1];
    acc = __builtin_amdgcn_wmma_f32_16x16x32_f16(false, af.v, false, bf.v,
                                                 (short)0, acc, false, false);
  }
  return acc;
}

__device__ inline void store_tile_f16(v8f acc, _Float16* dst, int strideH, int rowBase,
                                      int ct, const float* __restrict__ bias, bool relu) {
  int lane = threadIdx.x & 31;
  int n = lane & 15, hi = lane >> 4;
  float bv = bias[ct * 16 + n];
#pragma unroll
  for (int r = 0; r < 8; ++r) {
    float v = acc[r] + bv;
    if (relu) v = fmaxf(v, 0.0f);
    dst[(rowBase + r + 8 * hi) * strideH + ct * 16 + n] = (_Float16)v;
  }
}

__device__ inline void run_layer(const _Float16* src, int sStride,
                                 _Float16* dst, int dStride,
                                 const uint32_t* __restrict__ pw,
                                 const float* __restrict__ bias,
                                 int kChunks, int rowTiles, int colTiles, bool relu) {
  int wave = threadIdx.x >> 5;
  int nWaves = blockDim.x >> 5;
  int total = rowTiles * colTiles;
  for (int t = wave; t < total; t += nWaves) {
    int rt = t % rowTiles, ct = t / rowTiles;
    v8f acc = tile_acc((const uint32_t*)src, sStride, rt * 16, pw, kChunks, ct);
    store_tile_f16(acc, dst, dStride, rt * 16, ct, bias, relu);
  }
}

// ---------------------------------------------------------------------------
// Fused SA module: gather(x_j, rel-pos) -> 3-layer MLP (WMMA) -> masked max
// G queries per workgroup, 32 edges each.  256 threads (8 wave32).
// ---------------------------------------------------------------------------
template<int G, int CIN, int W0, int O0, int O1, int O2>
__global__ __launch_bounds__(256) void sa_mlp_kernel(
    const _Float16* __restrict__ xin,   // [B, Nsrc, CIN] f16
    const float*    __restrict__ posS,  // [B, Nsrc, 3]
    const float*    __restrict__ posQ,  // [B, M, 3]
    const int*      __restrict__ nbr,   // [B, M, 32]
    const unsigned* __restrict__ validm,// [B, M]
    const uint32_t* __restrict__ pw0, const float* __restrict__ bias0,
    const uint32_t* __restrict__ pw1, const float* __restrict__ bias1,
    const uint32_t* __restrict__ pw2, const float* __restrict__ bias2,
    _Float16* __restrict__ xout,        // [B, M, O2] f16
    int M, int Nsrc) {
  constexpr int ROWS = G * 32;
  constexpr int WA = W0;   // W0 > O1 for both modules
  constexpr int WB = O2;   // O2 >= O0 for both modules
  __shared__ __align__(16) _Float16 Abuf[ROWS * WA];
  __shared__ __align__(16) _Float16 Bbuf[ROWS * WB];
  __shared__ int s_nbr[ROWS];
  __shared__ unsigned s_val[G];
  __shared__ float s_qp[G * 3];

  int tid = threadIdx.x;
  int b = blockIdx.y;
  int qBase = blockIdx.x * G;

  if (tid < ROWS) s_nbr[tid] = nbr[(size_t)(b * M + qBase) * 32 + tid];
  if (tid < G)    s_val[tid] = validm[b * M + qBase + tid];
  if (tid < G * 3) s_qp[tid] = posQ[(size_t)(b * M + qBase) * 3 + tid];
  __syncthreads();

  // ---- gather layer-0 A matrix: [xj | rel | zero-pad] as f16
  {
    constexpr int TPR = 256 / ROWS;       // threads per row
    constexpr int CPT = W0 / TPR;         // cols per thread
    int row = tid / TPR, part = tid % TPR;
    int q = row >> 5;
    int src = s_nbr[row];
    const _Float16* xs = xin + ((size_t)b * Nsrc + src) * CIN;
    const float*    ps = posS + ((size_t)b * Nsrc + src) * 3;
    for (int c = part * CPT; c < (part + 1) * CPT; ++c) {
      _Float16 v;
      if (c < CIN)            v = xs[c];
      else if (c < CIN + 3)   v = (_Float16)(ps[c - CIN] - s_qp[q * 3 + (c - CIN)]);
      else                    v = (_Float16)0.0f;
      Abuf[row * WA + c] = v;
    }
  }
  __syncthreads();

  run_layer(Abuf, WA, Bbuf, WB, pw0, bias0, W0 / 32, ROWS / 16, O0 / 16, true);
  __syncthreads();
  run_layer(Bbuf, WB, Abuf, WA, pw1, bias1, O0 / 32, ROWS / 16, O1 / 16, true);
  __syncthreads();
  run_layer(Abuf, WA, Bbuf, WB, pw2, bias2, O1 / 32, ROWS / 16, O2 / 16, false);
  __syncthreads();

  // ---- masked max over the 32 edges of each query
  for (int idx = tid; idx < G * O2; idx += 256) {
    int q = idx / O2;
    int c = idx - q * O2;
    unsigned vm = s_val[q];
    float best = -INFINITY;
    for (int e = 0; e < 32; ++e)
      if ((vm >> e) & 1u) {
        float v = (float)Bbuf[(q * 32 + e) * WB + c];
        best = fmaxf(best, v);
      }
    xout[((size_t)(b * M + qBase + q)) * O2 + c] = (_Float16)best;
  }
}

// ---------------------------------------------------------------------------
// Head MLP: [x2 | pos2] (259 -> 256 -> 256 -> 256), f32 output for gridpool.
// 32 rows per workgroup, 256 threads.
// ---------------------------------------------------------------------------
__global__ __launch_bounds__(256) void mlp3_kernel(
    const _Float16* __restrict__ x2, const float* __restrict__ pos2,
    const uint32_t* __restrict__ pw0, const float* __restrict__ b0,
    const uint32_t* __restrict__ pw1, const float* __restrict__ b1,
    const uint32_t* __restrict__ pw2, const float* __restrict__ b2,
    float* __restrict__ h3, int M) {
  constexpr int ROWS = 32, WA = 288, WB = 256;
  __shared__ __align__(16) _Float16 Abuf[ROWS * WA];
  __shared__ __align__(16) _Float16 Bbuf[ROWS * WB];
  int tid = threadIdx.x;
  int b = blockIdx.y;
  int rBase = blockIdx.x * ROWS;

  {   // gather: 8 threads/row, 36 cols each
    int row = tid >> 3, part = tid & 7;
    int gr = rBase + row;
    const _Float16* xs = x2 + ((size_t)(b * M + gr)) * 256;
    const float*    pp = pos2 + ((size_t)(b * M + gr)) * 3;
    for (int c = part * 36; c < part * 36 + 36; ++c) {
      _Float16 v;
      if (c < 256)      v = xs[c];
      else if (c < 259) v = (_Float16)pp[c - 256];
      else              v = (_Float16)0.0f;
      Abuf[row * WA + c] = v;
    }
  }
  __syncthreads();
  run_layer(Abuf, WA, Bbuf, WB, pw0, b0, 9, 2, 16, true);
  __syncthreads();
  run_layer(Bbuf, WB, Abuf, WA, pw1, b1, 8, 2, 16, true);
  __syncthreads();
  {   // final layer -> global f32 (no relu)
    int wave = tid >> 5, lane = tid & 31;
    int n = lane & 15, hi = lane >> 4;
    for (int t = wave; t < 32; t += 8) {
      int rt = t % 2, ct = t / 2;
      v8f acc = tile_acc((const uint32_t*)Abuf, WA, rt * 16, pw2, 8, ct);
      float bv = b2[ct * 16 + n];
#pragma unroll
      for (int r = 0; r < 8; ++r) {
        int row = rBase + rt * 16 + r + 8 * hi;
        h3[((size_t)(b * M + row)) * 256 + ct * 16 + n] = acc[r] + bv;
      }
    }
  }
}

// ---------------------------------------------------------------------------
// Embedding lookup: x0[b,n,c] = f16(emb[polarity[b,n], c])
// ---------------------------------------------------------------------------
__global__ __launch_bounds__(256) void embed_kernel(const float* __restrict__ emb,
                                                    const int* __restrict__ pol,
                                                    _Float16* __restrict__ x0,
                                                    int totalPts) {
  int t = blockIdx.x * 256 + threadIdx.x;
  if (t >= totalPts * 64) return;
  int i = t >> 6, c = t & 63;
  x0[t] = (_Float16)emb[pol[i] * 64 + c];
}

// ---------------------------------------------------------------------------
// Iterative farthest point sampling: one workgroup per batch, dmin in LDS.
// ---------------------------------------------------------------------------
__global__ __launch_bounds__(256) void fps_kernel(const float* __restrict__ pos,
                                                  int n, int m, int* __restrict__ idx) {
  __shared__ float dmin[NN];
  __shared__ float rv[256];
  __shared__ int   ri[256];
  __shared__ float lastP[3];
  int tid = threadIdx.x;
  int b = blockIdx.x;
  const float* P = pos + (size_t)b * n * 3;
  for (int j = tid; j < n; j += 256) dmin[j] = INFINITY;
  if (tid == 0) {
    idx[(size_t)b * m] = 0;
    lastP[0] = P[0]; lastP[1] = P[1]; lastP[2] = P[2];
  }
  __syncthreads();
  for (int i = 1; i < m; ++i) {
    float lx = lastP[0], ly = lastP[1], lz = lastP[2];
    float bestv = -1.0f; int besti = n;
    for (int j = tid; j < n; j += 256) {
      float dx = P[3 * j] - lx, dy = P[3 * j + 1] - ly, dz = P[3 * j + 2] - lz;
      float d = dx * dx + dy * dy + dz * dz;
      float nd = fminf(dmin[j], d);
      dmin[j] = nd;
      if (nd > bestv || (nd == bestv && j < besti)) { bestv = nd; besti = j; }
    }
    rv[tid] = bestv; ri[tid] = besti;
    __syncthreads();
    for (int s = 128; s > 0; s >>= 1) {
      if (tid < s) {
        float ov = rv[tid + s]; int oi = ri[tid + s];
        if (ov > rv[tid] || (ov == rv[tid] && oi < ri[tid])) { rv[tid] = ov; ri[tid] = oi; }
      }
      __syncthreads();
    }
    if (tid == 0) {
      int bi = ri[0];
      idx[(size_t)b * m + i] = bi;
      lastP[0] = P[3 * bi]; lastP[1] = P[3 * bi + 1]; lastP[2] = P[3 * bi + 2];
    }
    __syncthreads();
  }
}

// ---------------------------------------------------------------------------
// Gather sampled positions
// ---------------------------------------------------------------------------
__global__ __launch_bounds__(256) void gather_pos_kernel(const float* __restrict__ pos,
                                                         const int* __restrict__ idx,
                                                         float* __restrict__ out,
                                                         int M, int Nsrc, int Btot) {
  int t = blockIdx.x * 256 + threadIdx.x;
  if (t >= Btot * M) return;
  int b = t / M;
  int s = idx[t];
  const float* p = pos + ((size_t)b * Nsrc + s) * 3;
  out[(size_t)t * 3 + 0] = p[0];
  out[(size_t)t * 3 + 1] = p[1];
  out[(size_t)t * 3 + 2] = p[2];
}

// ---------------------------------------------------------------------------
// Radius top-K (K=32 nearest within r) per query.  LDS-tiled source points,
// per-thread insertion list (scratch).  Valid bitmask per query.
// ---------------------------------------------------------------------------
__global__ __launch_bounds__(128) void knn_kernel(const float* __restrict__ posQ,
                                                  const float* __restrict__ posS,
                                                  int Mq, int Ns, float r2,
                                                  int* __restrict__ nbr,
                                                  unsigned* __restrict__ validm) {
  __shared__ float sp[256 * 3];
  int tid = threadIdx.x;
  int b = blockIdx.y;
  int q = blockIdx.x * 128 + tid;
  const float* qp = posQ + ((size_t)(b * Mq + q)) * 3;
  float qx = qp[0], qy = qp[1], qz = qp[2];
  float bd[32]; int bix[32];
#pragma unroll
  for (int k = 0; k < 32; ++k) { bd[k] = 3.0e38f; bix[k] = 0; }
  for (int base = 0; base < Ns; base += 256) {
    for (int t = tid; t < 768; t += 128)
      sp[t] = posS[((size_t)b * Ns + base) * 3 + t];
    __syncthreads();
    for (int jj = 0; jj < 256; ++jj) {
      float dx = sp[jj * 3] - qx, dy = sp[jj * 3 + 1] - qy, dz = sp[jj * 3 + 2] - qz;
      float d2 = dx * dx + dy * dy + dz * dz;
      if (d2 <= r2 && d2 < bd[31]) {
        int p = 31;
        while (p > 0 && bd[p - 1] > d2) { bd[p] = bd[p - 1]; bix[p] = bix[p - 1]; --p; }
        bd[p] = d2; bix[p] = base + jj;
      }
    }
    __syncthreads();
  }
  unsigned vm = 0;
  size_t o = ((size_t)(b * Mq + q)) * 32;
  for (int k = 0; k < 32; ++k) {
    if (bd[k] <= r2) { vm |= (1u << k); nbr[o + k] = bix[k]; }
    else             { nbr[o + k] = 0; }
  }
  validm[b * Mq + q] = vm;
}

// ---------------------------------------------------------------------------
// Grid cell id + voxel max pool
// ---------------------------------------------------------------------------
__global__ __launch_bounds__(256) void cellid_kernel(const float* __restrict__ pos2,
                                                     int* __restrict__ cid,
                                                     int M, int Btot) {
  int t = blockIdx.x * 256 + threadIdx.x;
  if (t >= Btot * M) return;
  float x = pos2[(size_t)t * 3 + 0];
  float y = pos2[(size_t)t * 3 + 1];
  int cx = (int)floorf(x * (1.0f / 16.0f));
  int cy = (int)floorf(y * (1.0f / 16.0f));
  cx = min(max(cx, 0), NCOLS - 1);
  cy = min(max(cy, 0), NROWS - 1);
  cid[t] = cy * NCOLS + cx;
}

__global__ __launch_bounds__(256) void gridpool_kernel(const float* __restrict__ h3,
                                                       const int* __restrict__ cid,
                                                       float* __restrict__ out, int M) {
  __shared__ int sc[MM2];
  int tid = threadIdx.x;           // channel (256)
  int cell = blockIdx.x;
  int b = blockIdx.y;
  for (int t = tid; t < M; t += 256) sc[t] = cid[b * M + t];
  __syncthreads();
  float acc = -INFINITY;
  for (int p = 0; p < M; ++p)
    if (sc[p] == cell)
      acc = fmaxf(acc, h3[((size_t)(b * M + p)) * 256 + tid]);
  if (!(acc > -3.0e38f)) acc = 0.0f;   // empty cell -> 0
  out[((size_t)(b * NCELLS + cell)) * 256 + tid] = acc;
}

// ---------------------------------------------------------------------------
// Host launcher
// ---------------------------------------------------------------------------
extern "C" void kernel_launch(void* const* d_in, const int* in_sizes, int n_in,
                              void* d_out, int out_size, void* d_ws, size_t ws_size,
                              hipStream_t stream) {
  const float* pos = (const float*)d_in[0];
  const float* emb = (const float*)d_in[1];
  const float* w1a = (const float*)d_in[2];  const float* b1a = (const float*)d_in[3];
  const float* w1b = (const float*)d_in[4];  const float* b1b = (const float*)d_in[5];
  const float* w1c = (const float*)d_in[6];  const float* b1c = (const float*)d_in[7];
  const float* w2a = (const float*)d_in[8];  const float* b2a = (const float*)d_in[9];
  const float* w2b = (const float*)d_in[10]; const float* b2b = (const float*)d_in[11];
  const float* w2c = (const float*)d_in[12]; const float* b2c = (const float*)d_in[13];
  const float* w3a = (const float*)d_in[14]; const float* b3a = (const float*)d_in[15];
  const float* w3b = (const float*)d_in[16]; const float* b3b = (const float*)d_in[17];
  const float* w3c = (const float*)d_in[18]; const float* b3c = (const float*)d_in[19];
  const int* polarity = (const int*)d_in[20];
  float* out = (float*)d_out;

  char* ws = (char*)d_ws;
  size_t off = 0;
  auto alloc = [&](size_t bytes) -> void* {
    void* p = ws + off;
    off += (bytes + 255) & ~(size_t)255;
    return p;
  };
  _Float16* x0  = (_Float16*)alloc((size_t)BB * NN * DD1 * 2);
  int*      idx1 = (int*)alloc((size_t)BB * MM1 * 4);
  float*    pos1 = (float*)alloc((size_t)BB * MM1 * 3 * 4);
  int*      nbr1 = (int*)alloc((size_t)BB * MM1 * KK * 4);
  unsigned* val1 = (unsigned*)alloc((size_t)BB * MM1 * 4);
  _Float16* x1  = (_Float16*)alloc((size_t)BB * MM1 * DD2 * 2);
  int*      idx2 = (int*)alloc((size_t)BB * MM2 * 4);
  float*    pos2 = (float*)alloc((size_t)BB * MM2 * 3 * 4);
  int*      nbr2 = (int*)alloc((size_t)BB * MM2 * KK * 4);
  unsigned* val2 = (unsigned*)alloc((size_t)BB * MM2 * 4);
  _Float16* x2  = (_Float16*)alloc((size_t)BB * MM2 * DD3 * 2);
  float*    h3  = (float*)alloc((size_t)BB * MM2 * DD3 * 4);
  int*      cid = (int*)alloc((size_t)BB * MM2 * 4);

  auto packAlloc = [&](int fin, int fout) -> uint32_t* {
    int kc = (fin + 31) / 32, ct = fout / 16;
    return (uint32_t*)alloc((size_t)ct * kc * 32 * 8 * 4);
  };
  uint32_t* pw1a = packAlloc(67, 64);
  uint32_t* pw1b = packAlloc(64, 64);
  uint32_t* pw1c = packAlloc(64, 128);
  uint32_t* pw2a = packAlloc(131, 128);
  uint32_t* pw2b = packAlloc(128, 128);
  uint32_t* pw2c = packAlloc(128, 256);
  uint32_t* pw3a = packAlloc(259, 256);
  uint32_t* pw3b = packAlloc(256, 256);
  uint32_t* pw3c = packAlloc(256, 256);

  auto packLaunch = [&](const float* W, int fin, int fout, uint32_t* dst) {
    int kc = (fin + 31) / 32, ct = fout / 16;
    int total = ct * kc * 32;
    pack_w_kernel<<<(total + 255) / 256, 256, 0, stream>>>(W, fin, fout, dst, kc);
  };
  packLaunch(w1a, 67, 64, pw1a);   packLaunch(w1b, 64, 64, pw1b);   packLaunch(w1c, 64, 128, pw1c);
  packLaunch(w2a, 131, 128, pw2a); packLaunch(w2b, 128, 128, pw2b); packLaunch(w2c, 128, 256, pw2c);
  packLaunch(w3a, 259, 256, pw3a); packLaunch(w3b, 256, 256, pw3b); packLaunch(w3c, 256, 256, pw3c);

  // embedding lookup
  embed_kernel<<<(BB * NN * 64 + 255) / 256, 256, 0, stream>>>(emb, polarity, x0, BB * NN);

  // ---- SA module 1
  fps_kernel<<<BB, 256, 0, stream>>>(pos, NN, MM1, idx1);
  gather_pos_kernel<<<(BB * MM1 + 255) / 256, 256, 0, stream>>>(pos, idx1, pos1, MM1, NN, BB);
  knn_kernel<<<dim3(MM1 / 128, BB), 128, 0, stream>>>(pos1, pos, MM1, NN, 32.0f * 32.0f, nbr1, val1);
  sa_mlp_kernel<4, 64, 96, 64, 64, 128><<<dim3(MM1 / 4, BB), 256, 0, stream>>>(
      x0, pos, pos1, nbr1, val1, pw1a, b1a, pw1b, b1b, pw1c, b1c, x1, MM1, NN);

  // ---- SA module 2
  fps_kernel<<<BB, 256, 0, stream>>>(pos1, MM1, MM2, idx2);
  gather_pos_kernel<<<(BB * MM2 + 255) / 256, 256, 0, stream>>>(pos1, idx2, pos2, MM2, MM1, BB);
  knn_kernel<<<dim3(MM2 / 128, BB), 128, 0, stream>>>(pos2, pos1, MM2, MM1, 64.0f * 64.0f, nbr2, val2);
  sa_mlp_kernel<2, 128, 160, 128, 128, 256><<<dim3(MM2 / 2, BB), 256, 0, stream>>>(
      x1, pos1, pos2, nbr2, val2, pw2a, b2a, pw2b, b2b, pw2c, b2c, x2, MM2, MM1);

  // ---- head MLP + grid pooling
  mlp3_kernel<<<dim3(MM2 / 32, BB), 256, 0, stream>>>(x2, pos2, pw3a, b3a, pw3b, b3b, pw3c, b3c, h3, MM2);
  cellid_kernel<<<(BB * MM2 + 255) / 256, 256, 0, stream>>>(pos2, cid, MM2, BB);
  gridpool_kernel<<<dim3(NCELLS, BB), 256, 0, stream>>>(h3, cid, out, MM2);

  (void)in_sizes; (void)n_in; (void)out_size; (void)ws_size;
}
